// CausalAttention_40845138985324
// MI455X (gfx1250) — matrix operand order repaired and checked by reading, not compiled, not run on hardware
//
#include <hip/hip_runtime.h>
#include <math.h>

// Problem constants (match reference)
#define T_SEQ   2048
#define D_MODEL 2048
#define H_HEADS 16
#define HD      128
#define L_LAT   512
#define RD      64
#define K_TOP   32

typedef __attribute__((ext_vector_type(16))) __bf16 v16bf;
typedef __attribute__((ext_vector_type(8)))  __bf16 v8bf;
typedef __attribute__((ext_vector_type(8)))  float  v8f;

// ---------------------------------------------------------------------------
// f32 -> bf16 round-to-nearest-even (bit manipulation; avoids relying on
// scalar __bf16 conversion support)
// ---------------------------------------------------------------------------
__device__ __forceinline__ unsigned short f32_to_bf16(float f) {
  union { float f; unsigned int u; } c; c.f = f;
  unsigned int u = c.u;
  if ((u & 0x7fffffffu) > 0x7f800000u) return 0x7fc0;   // NaN
  u += 0x7fffu + ((u >> 16) & 1u);                      // RNE
  return (unsigned short)(u >> 16);
}

// ---------------------------------------------------------------------------
// Elementwise pack: f32 row-major -> bf16 row-major (WMMA A operand)
// ---------------------------------------------------------------------------
__global__ void pack_bf16_kernel(const float* __restrict__ in,
                                 unsigned short* __restrict__ out, int n) {
  int i = blockIdx.x * 256 + threadIdx.x;
  if (i < n) out[i] = f32_to_bf16(in[i]);
}

// ---------------------------------------------------------------------------
// Pack + transpose: W (Kdim x N, f32 row-major) -> Bt (Npad x Kdim, bf16),
// zero-padding rows N..Npad-1 so GEMM tiles are always full.
// ---------------------------------------------------------------------------
__global__ void pack_bf16_transpose_kernel(const float* __restrict__ W,
                                           unsigned short* __restrict__ Bt,
                                           int Kdim, int N, int Npad) {
  int i = blockIdx.x * 256 + threadIdx.x;
  if (i >= Npad * Kdim) return;
  int n = i / Kdim;
  int k = i - n * Kdim;
  Bt[i] = (n < N) ? f32_to_bf16(W[(size_t)k * N + n]) : (unsigned short)0;
}

// ---------------------------------------------------------------------------
// bf16 WMMA GEMM:  C(f32, MxN) = A(bf16, MxK row-major) * Bt^T(bf16, Npad x K)
// Block = 256 threads = 8 waves (wave32). Block tile 128x128, wave tile 32x64
// (2 M-subtiles x 4 N-subtiles of 16x16), K-step 32.
// Operand layouts (contiguous k order, same for A and B):
//   A 16x32 bf16: lane m=lane%16; element e -> K = 16*(lane>>4) + e
//       => two adjacent 8-elem (16B) chunks at k0+16*half and k0+16*half+8.
//   B 32x16 bf16: lane n=lane%16; element e -> K = 16*(lane>>4) + e
//       => one contiguous 16-elem (32B) chunk of the transposed weight row.
//   C 16x16 f32: vgpr r -> row = r + 8*(lane>>4), col = lane%16.
// ---------------------------------------------------------------------------
__global__ __launch_bounds__(256)
void gemm_bf16_wmma_kernel(const unsigned short* __restrict__ A,
                           const unsigned short* __restrict__ Bt,
                           float* __restrict__ C,
                           int M, int N, int Kdim) {
  const int lane = threadIdx.x & 31;
  const int wave = threadIdx.x >> 5;
  const int wm   = wave & 3;      // 4 waves along M
  const int wn   = wave >> 2;     // 2 waves along N
  const int half = lane >> 4;
  const int l16  = lane & 15;

  const int m_base = blockIdx.y * 128 + wm * 32;
  const int n_base = blockIdx.x * 128 + wn * 64;

  const __bf16* Ab = reinterpret_cast<const __bf16*>(A);
  const __bf16* Bb = reinterpret_cast<const __bf16*>(Bt);

  v8f acc[2][4];
#pragma unroll
  for (int mi = 0; mi < 2; ++mi)
#pragma unroll
    for (int ni = 0; ni < 4; ++ni)
#pragma unroll
      for (int r = 0; r < 8; ++r) acc[mi][ni][r] = 0.0f;

  const __bf16* aRow[2];
#pragma unroll
  for (int mi = 0; mi < 2; ++mi)
    aRow[mi] = Ab + (size_t)(m_base + mi * 16 + l16) * Kdim + 16 * half;
  const __bf16* bRow[4];
#pragma unroll
  for (int ni = 0; ni < 4; ++ni)
    bRow[ni] = Bb + (size_t)(n_base + ni * 16 + l16) * Kdim + 16 * half;

  for (int k0 = 0; k0 < Kdim; k0 += 32) {
    v16bf afrag[2], bfrag[4];
#pragma unroll
    for (int mi = 0; mi < 2; ++mi) {
      v8bf lo = *reinterpret_cast<const v8bf*>(aRow[mi] + k0);
      v8bf hi = *reinterpret_cast<const v8bf*>(aRow[mi] + k0 + 8);
      afrag[mi] = __builtin_shufflevector(lo, hi,
          0, 1, 2, 3, 4, 5, 6, 7, 8, 9, 10, 11, 12, 13, 14, 15);
    }
#pragma unroll
    for (int ni = 0; ni < 4; ++ni)
      bfrag[ni] = *reinterpret_cast<const v16bf*>(bRow[ni] + k0);

    if (k0 + 32 < Kdim) {  // gfx1250 global_prefetch_b8 for next k-step
      __builtin_prefetch(aRow[0] + k0 + 32, 0, 1);
      __builtin_prefetch(bRow[0] + k0 + 32, 0, 1);
    }

#pragma unroll
    for (int mi = 0; mi < 2; ++mi)
#pragma unroll
      for (int ni = 0; ni < 4; ++ni)
        acc[mi][ni] = __builtin_amdgcn_wmma_f32_16x16x32_bf16(
            false, afrag[mi], false, bfrag[ni],
            (short)0, acc[mi][ni], false, false);
  }

#pragma unroll
  for (int mi = 0; mi < 2; ++mi)
#pragma unroll
    for (int ni = 0; ni < 4; ++ni) {
      int col = n_base + ni * 16 + l16;
      if (col < N) {
#pragma unroll
        for (int r = 0; r < 8; ++r) {
          int row = m_base + mi * 16 + r + 8 * half;
          C[(size_t)row * N + col] = acc[mi][ni][r];
        }
      }
    }
}

// ---------------------------------------------------------------------------
// In-place RoPE over last dim rd (pairs (j, j+rd/2) share angle t*theta^(-j/(rd/2)))
// Layout: p[t][h][rd], one thread per (t,h,j).
// ---------------------------------------------------------------------------
__global__ void rope_kernel(float* __restrict__ p, int T, int nH, int rd) {
  int halfrd = rd >> 1;
  int total = T * nH * halfrd;
  int i = blockIdx.x * 256 + threadIdx.x;
  if (i >= total) return;
  int j  = i % halfrd;
  int th = i / halfrd;
  int h  = th % nH;
  int t  = th / nH;

  float freq = expf(-(float)j * (logf(10000.0f) / (float)halfrd));
  float ang = (float)t * freq;
  float s, c;
  sincosf(ang, &s, &c);

  size_t base = ((size_t)t * nH + h) * rd;
  float a = p[base + j];
  float b = p[base + j + halfrd];
  p[base + j]          = a * c - b * s;
  p[base + j + halfrd] = b * c + a * s;
}

// ---------------------------------------------------------------------------
// Top-k gather attention. One wave32 per (t, h); lane kk owns selected token kk.
// score_kk = (q[t,h,:]·k[idx,h,:] + qr[t,h,:]·kr[idx,:]) / sqrt(HD+RD)
// softmax across the 32 lanes via shuffle reductions, then each lane
// accumulates 4 output dims with shuffle-broadcast weights.
// ---------------------------------------------------------------------------
__global__ __launch_bounds__(256)
void topk_attention_kernel(const float* __restrict__ q,    // T x (H*HD)
                           const float* __restrict__ qr,   // T x (H*RD)
                           const float* __restrict__ kmat, // T x (H*HD)
                           const float* __restrict__ vmat, // T x (H*HD)
                           const float* __restrict__ kr,   // T x RD
                           const int*   __restrict__ topk, // T x K
                           float* __restrict__ out) {      // T x (H*HD)
  __shared__ float sq[8][HD];
  __shared__ float sqr[8][RD];
  const int lane = threadIdx.x & 31;
  const int wv   = threadIdx.x >> 5;
  const int w    = blockIdx.x * 8 + wv;
  const int t    = w / H_HEADS;
  const int h    = w % H_HEADS;

  const float* qrow  = q  + (size_t)t * D_MODEL + h * HD;
  const float* qrrow = qr + (size_t)t * (H_HEADS * RD) + h * RD;

  *reinterpret_cast<float4*>(&sq[wv][lane * 4]) =
      *reinterpret_cast<const float4*>(qrow + lane * 4);
  float2 t2 = *reinterpret_cast<const float2*>(qrrow + lane * 2);
  sqr[wv][lane * 2]     = t2.x;
  sqr[wv][lane * 2 + 1] = t2.y;
  __syncthreads();

  int idx = topk[(size_t)t * K_TOP + lane];
  const float* krow  = kmat + (size_t)idx * D_MODEL + h * HD;
  const float* krrow = kr   + (size_t)idx * RD;

  float score = 0.f;
#pragma unroll
  for (int d = 0; d < HD; d += 4) {
    float4 kv = *reinterpret_cast<const float4*>(krow + d);
    score += sq[wv][d] * kv.x + sq[wv][d + 1] * kv.y +
             sq[wv][d + 2] * kv.z + sq[wv][d + 3] * kv.w;
  }
#pragma unroll
  for (int r = 0; r < RD; r += 4) {
    float4 kv = *reinterpret_cast<const float4*>(krrow + r);
    score += sqr[wv][r] * kv.x + sqr[wv][r + 1] * kv.y +
             sqr[wv][r + 2] * kv.z + sqr[wv][r + 3] * kv.w;
  }
  score *= 0.07216878364870322f;  // 1/sqrt(HD+RD) = 1/sqrt(192)

  float m = score;
#pragma unroll
  for (int off = 16; off >= 1; off >>= 1) m = fmaxf(m, __shfl_xor(m, off, 32));
  float e = __expf(score - m);
  float ssum = e;
#pragma unroll
  for (int off = 16; off >= 1; off >>= 1) ssum += __shfl_xor(ssum, off, 32);
  float attn = e / ssum;

  float4 accv = make_float4(0.f, 0.f, 0.f, 0.f);
#pragma unroll
  for (int kk = 0; kk < K_TOP; ++kk) {
    float a  = __shfl(attn, kk, 32);
    int   id = __shfl(idx,  kk, 32);
    float4 vv = *reinterpret_cast<const float4*>(
        vmat + (size_t)id * D_MODEL + h * HD + lane * 4);
    accv.x += a * vv.x; accv.y += a * vv.y;
    accv.z += a * vv.z; accv.w += a * vv.w;
  }
  *reinterpret_cast<float4*>(out + (size_t)t * D_MODEL + h * HD + lane * 4) = accv;
}

// ---------------------------------------------------------------------------
// Host-side orchestration (graph-capture safe: only kernel launches on stream)
// ---------------------------------------------------------------------------
extern "C" void kernel_launch(void* const* d_in, const int* in_sizes, int n_in,
                              void* d_out, int out_size, void* d_ws, size_t ws_size,
                              hipStream_t stream) {
  const float* x    = (const float*)d_in[0];
  const float* Wqd  = (const float*)d_in[1];
  const float* Wqu  = (const float*)d_in[2];
  const float* Wqr  = (const float*)d_in[3];
  const float* Wkvd = (const float*)d_in[4];
  const float* Wku  = (const float*)d_in[5];
  const float* Wvu  = (const float*)d_in[6];
  const float* Wkr  = (const float*)d_in[7];
  const float* Wo   = (const float*)d_in[8];
  const int*   topk = (const int*)d_in[9];
  float* out = (float*)d_out;

  char* ws = (char*)d_ws;
  size_t off = 0;
  auto alloc = [&](size_t bytes) -> void* {
    off = (off + 255) & ~(size_t)255;
    void* p = ws + off;
    off += bytes;
    return p;
  };

  // bf16 packs (unsigned short storage)
  unsigned short* xb    = (unsigned short*)alloc((size_t)T_SEQ * D_MODEL * 2);
  unsigned short* WqdT  = (unsigned short*)alloc((size_t)L_LAT * D_MODEL * 2);   // 512 x 2048
  unsigned short* WquT  = (unsigned short*)alloc((size_t)D_MODEL * L_LAT * 2);   // 2048 x 512
  unsigned short* WqrT  = (unsigned short*)alloc((size_t)1024 * L_LAT * 2);      // 1024 x 512
  unsigned short* WkvdT = (unsigned short*)alloc((size_t)L_LAT * D_MODEL * 2);
  unsigned short* WkuT  = (unsigned short*)alloc((size_t)D_MODEL * L_LAT * 2);
  unsigned short* WvuT  = (unsigned short*)alloc((size_t)D_MODEL * L_LAT * 2);
  unsigned short* WkrT  = (unsigned short*)alloc((size_t)128 * D_MODEL * 2);     // 64->128 padded
  unsigned short* WoT   = (unsigned short*)alloc((size_t)D_MODEL * D_MODEL * 2);
  unsigned short* qlb   = (unsigned short*)alloc((size_t)T_SEQ * L_LAT * 2);
  unsigned short* ckvb  = (unsigned short*)alloc((size_t)T_SEQ * L_LAT * 2);
  unsigned short* aob   = (unsigned short*)alloc((size_t)T_SEQ * D_MODEL * 2);
  // f32 intermediates
  float* qlf  = (float*)alloc((size_t)T_SEQ * L_LAT * 4);
  float* ckvf = (float*)alloc((size_t)T_SEQ * L_LAT * 4);
  float* qf   = (float*)alloc((size_t)T_SEQ * D_MODEL * 4);
  float* qrf  = (float*)alloc((size_t)T_SEQ * 1024 * 4);
  float* kf   = (float*)alloc((size_t)T_SEQ * D_MODEL * 4);
  float* vf   = (float*)alloc((size_t)T_SEQ * D_MODEL * 4);
  float* krf  = (float*)alloc((size_t)T_SEQ * RD * 4);
  float* aof  = (float*)alloc((size_t)T_SEQ * D_MODEL * 4);

  const dim3 blk(256);
  auto g1 = [](int n) { return dim3((n + 255) / 256); };

  // --- pack inputs to bf16 (weights transposed for WMMA B layout) ---
  pack_bf16_kernel<<<g1(T_SEQ * D_MODEL), blk, 0, stream>>>(x, xb, T_SEQ * D_MODEL);
  pack_bf16_transpose_kernel<<<g1(512 * 2048), blk, 0, stream>>>(Wqd,  WqdT,  2048, 512, 512);
  pack_bf16_transpose_kernel<<<g1(2048 * 512), blk, 0, stream>>>(Wqu,  WquT,  512, 2048, 2048);
  pack_bf16_transpose_kernel<<<g1(1024 * 512), blk, 0, stream>>>(Wqr,  WqrT,  512, 1024, 1024);
  pack_bf16_transpose_kernel<<<g1(512 * 2048), blk, 0, stream>>>(Wkvd, WkvdT, 2048, 512, 512);
  pack_bf16_transpose_kernel<<<g1(2048 * 512), blk, 0, stream>>>(Wku,  WkuT,  512, 2048, 2048);
  pack_bf16_transpose_kernel<<<g1(2048 * 512), blk, 0, stream>>>(Wvu,  WvuT,  512, 2048, 2048);
  pack_bf16_transpose_kernel<<<g1(128 * 2048), blk, 0, stream>>>(Wkr,  WkrT,  2048, 64, 128);
  pack_bf16_transpose_kernel<<<g1(2048 * 2048), blk, 0, stream>>>(Wo,  WoT,   2048, 2048, 2048);

  // --- latent projections ---
  gemm_bf16_wmma_kernel<<<dim3(512 / 128, T_SEQ / 128), blk, 0, stream>>>(
      xb, WqdT, qlf, T_SEQ, 512, 2048);                 // ql = x @ Wqd
  gemm_bf16_wmma_kernel<<<dim3(512 / 128, T_SEQ / 128), blk, 0, stream>>>(
      xb, WkvdT, ckvf, T_SEQ, 512, 2048);               // ckv = x @ Wkvd
  pack_bf16_kernel<<<g1(T_SEQ * L_LAT), blk, 0, stream>>>(qlf,  qlb,  T_SEQ * L_LAT);
  pack_bf16_kernel<<<g1(T_SEQ * L_LAT), blk, 0, stream>>>(ckvf, ckvb, T_SEQ * L_LAT);

  // --- up-projections ---
  gemm_bf16_wmma_kernel<<<dim3(2048 / 128, T_SEQ / 128), blk, 0, stream>>>(
      qlb, WquT, qf, T_SEQ, 2048, 512);                 // q  = ql @ Wqu
  gemm_bf16_wmma_kernel<<<dim3(1024 / 128, T_SEQ / 128), blk, 0, stream>>>(
      qlb, WqrT, qrf, T_SEQ, 1024, 512);                // qr = ql @ Wqr
  gemm_bf16_wmma_kernel<<<dim3(2048 / 128, T_SEQ / 128), blk, 0, stream>>>(
      ckvb, WkuT, kf, T_SEQ, 2048, 512);                // k  = ckv @ Wku
  gemm_bf16_wmma_kernel<<<dim3(2048 / 128, T_SEQ / 128), blk, 0, stream>>>(
      ckvb, WvuT, vf, T_SEQ, 2048, 512);                // v  = ckv @ Wvu
  gemm_bf16_wmma_kernel<<<dim3(128 / 128, T_SEQ / 128), blk, 0, stream>>>(
      xb, WkrT, krf, T_SEQ, 64, 2048);                  // kr = x @ Wkr

  // --- RoPE (in place) ---
  rope_kernel<<<g1(T_SEQ * H_HEADS * (RD / 2)), blk, 0, stream>>>(qrf, T_SEQ, H_HEADS, RD);
  rope_kernel<<<g1(T_SEQ * 1 * (RD / 2)), blk, 0, stream>>>(krf, T_SEQ, 1, RD);

  // --- top-k sparse attention (one wave per (t,h)) ---
  topk_attention_kernel<<<dim3(T_SEQ * H_HEADS / 8), blk, 0, stream>>>(
      qf, qrf, kf, vf, krf, topk, aof);

  // --- output projection ---
  pack_bf16_kernel<<<g1(T_SEQ * D_MODEL), blk, 0, stream>>>(aof, aob, T_SEQ * D_MODEL);
  gemm_bf16_wmma_kernel<<<dim3(2048 / 128, T_SEQ / 128), blk, 0, stream>>>(
      aob, WoT, out, T_SEQ, 2048, 2048);                // out = ao @ Wo
}
